// LinearWithLoRA_42030549958693
// MI455X (gfx1250) — matrix operand order repaired and checked by reading, not compiled, not run on hardware
//
#include <hip/hip_runtime.h>

typedef __attribute__((ext_vector_type(16))) _Float16 v16h;
typedef __attribute__((ext_vector_type(8)))  float    v8f;

#define IN_DIM   4096
#define OUT_DIM  4096
#define M_DIM    8192   // B*S = 4*2048
#define GS       64
#define RANK     16
#define LORA_SCALE 2.0f // alpha/rank = 32/16

#define BM 128
#define BN 256
#define BK 32
#define NK (IN_DIM / BK)   // 128 K-steps
#define LDSK 40            // 32 + 8 halves pad -> 80B row stride (16B aligned)

union Frag16 { v16h v; uint4 q[2]; };

// ---------------------------------------------------------------------------
// Pass 1: Weff[n][k] = f16( wq[n][k]*scale[n*64 + k/64] + 2*sum_r A[k][r]*B[r][n] )
// ---------------------------------------------------------------------------
__global__ void lora_wconv_kernel(const int* __restrict__ wq,
                                  const float* __restrict__ scale,
                                  const float* __restrict__ A,
                                  const float* __restrict__ Blora,
                                  _Float16* __restrict__ Weff) {
    __shared__ float Bn[RANK];
    const int n = blockIdx.x;
    const int t = threadIdx.x;
    if (t < RANK) Bn[t] = LORA_SCALE * Blora[t * OUT_DIM + n];
    __syncthreads();

    const int*   wrow = wq    + (size_t)n * IN_DIM;
    const float* srow = scale + (size_t)n * (IN_DIM / GS);
    _Float16*    orow = Weff  + (size_t)n * IN_DIM;

    for (int kk = 0; kk < IN_DIM / 256; ++kk) {
        const int k = kk * 256 + t;
        float acc = (float)wrow[k] * srow[k >> 6];
        const float* ar = A + (size_t)k * RANK;
        #pragma unroll
        for (int r = 0; r < RANK; ++r) acc += ar[r] * Bn[r];
        orow[k] = (_Float16)acc;
    }
}

// ---------------------------------------------------------------------------
// Pass 2: x fp32 -> f16 (8 elements / thread, vectorized)
// ---------------------------------------------------------------------------
__global__ void xconv_kernel(const float* __restrict__ x,
                             _Float16* __restrict__ xh) {
    const size_t i = ((size_t)blockIdx.x * 256 + threadIdx.x) * 8;
    const float4 a = *(const float4*)(x + i);
    const float4 b = *(const float4*)(x + i + 4);
    union { _Float16 h[8]; uint4 q; } o;
    o.h[0] = (_Float16)a.x; o.h[1] = (_Float16)a.y;
    o.h[2] = (_Float16)a.z; o.h[3] = (_Float16)a.w;
    o.h[4] = (_Float16)b.x; o.h[5] = (_Float16)b.y;
    o.h[6] = (_Float16)b.z; o.h[7] = (_Float16)b.w;
    *(uint4*)(xh + i) = o.q;
}

// ---------------------------------------------------------------------------
// Async global -> LDS helpers (CDNA5 ASYNCcnt path)
// ---------------------------------------------------------------------------
__device__ __forceinline__ unsigned lds_addr32(const void* p) {
    // ISA 10.2: flat addresses in the LDS aperture use addr[31:0] as the LDS
    // byte address — truncating the generic pointer yields the DS address.
    return (unsigned)(unsigned long long)p;
}

// A: 16 halves (2x b128) per thread.  B: 32 halves (4x b128) per thread.
// INST_OFFSET applies to both LDS and global addresses (ISA 10.7 pseudocode),
// so each LDS/global address pair serves all its b128 slices.
__device__ __forceinline__ void async_tile_load(unsigned alds, unsigned blds,
                                                const _Float16* ag,
                                                const _Float16* bg) {
    asm volatile(
        "global_load_async_to_lds_b128 %0, %2, off\n\t"
        "global_load_async_to_lds_b128 %0, %2, off offset:16\n\t"
        "global_load_async_to_lds_b128 %1, %3, off\n\t"
        "global_load_async_to_lds_b128 %1, %3, off offset:16\n\t"
        "global_load_async_to_lds_b128 %1, %3, off offset:32\n\t"
        "global_load_async_to_lds_b128 %1, %3, off offset:48"
        :: "v"(alds), "v"(blds), "v"(ag), "v"(bg)
        : "memory");
}

__device__ __forceinline__ void wait_async0() {
    asm volatile("s_wait_asynccnt 0x0" ::: "memory");
}

// ---------------------------------------------------------------------------
// Main GEMM: out[8192][4096] = Xh[8192][4096] * Weff[4096][4096]^T
// 128x256 block tile, BK=32, double-buffered LDS filled by async DMA.
// 8 wave32 waves as 2(m) x 4(n); each wave = 64x64 via 4x4 WMMA fragments
// (16 ds_load_b128 per 16 WMMA -> 1.0 LDS-load/WMMA ratio).
// ---------------------------------------------------------------------------
__device__ __forceinline__ void compute_tile(const _Float16 (*__restrict__ As)[LDSK],
                                             const _Float16 (*__restrict__ Bs)[LDSK],
                                             v8f acc[4][4],
                                             int wm, int wn, int r16, int hi) {
    // B fragments first: live across the whole mt loop
    Frag16 b[4];
    #pragma unroll
    for (int nt = 0; nt < 4; ++nt) {
        const _Float16* p = &Bs[wn + nt * 16 + r16][hi * 16];
        b[nt].q[0] = *(const uint4*)(p);
        b[nt].q[1] = *(const uint4*)(p + 8);
    }
    #pragma unroll
    for (int mt = 0; mt < 4; ++mt) {
        Frag16 a;
        const _Float16* p = &As[wm + mt * 16 + r16][hi * 16];
        a.q[0] = *(const uint4*)(p);
        a.q[1] = *(const uint4*)(p + 8);
        #pragma unroll
        for (int nt = 0; nt < 4; ++nt)
            acc[mt][nt] = __builtin_amdgcn_wmma_f32_16x16x32_f16(
                false, a.v, false, b[nt].v,
                (short)0, acc[mt][nt], false, false);
    }
}

__global__ void __launch_bounds__(256)
lora_gemm_kernel(const _Float16* __restrict__ Xh,
                 const _Float16* __restrict__ Wh,
                 float* __restrict__ out) {
    __shared__ _Float16 As[2][BM][LDSK];   // 2 x 10 KB
    __shared__ _Float16 Bs[2][BN][LDSK];   // 2 x 20 KB

    const int tid  = threadIdx.x;
    const int lane = tid & 31;
    const int wid  = tid >> 5;          // 0..7
    const int wm   = (wid >> 2) * 64;   // wave m offset in block: 0/64
    const int wn   = (wid & 3)  * 64;   // wave n offset in block: 0..192
    const int r16  = lane & 15;
    const int hi   = lane >> 4;         // half-wave select

    const size_t m0 = (size_t)blockIdx.y * BM;
    const size_t n0 = (size_t)blockIdx.x * BN;

    // A loader: thread -> (row tid/2, 16-half slab); B loader: thread -> full row tid
    const int arow = tid >> 1;
    const int acol = (tid & 1) * 16;
    const _Float16* ag = Xh + (m0 + arow) * IN_DIM + acol;
    const _Float16* bg = Wh + (n0 + tid) * IN_DIM;

    const unsigned alds[2] = { lds_addr32(&As[0][arow][acol]),
                               lds_addr32(&As[1][arow][acol]) };
    const unsigned blds[2] = { lds_addr32(&Bs[0][tid][0]),
                               lds_addr32(&Bs[1][tid][0]) };

    v8f acc[4][4] = {};

    // prologue: DMA first tile into buffer 0
    async_tile_load(alds[0], blds[0], ag, bg);

    for (int it = 0; it < NK; it += 2) {
        // ---- phase 0: consume buf0, DMA k+1 into buf1 under the WMMAs ----
        wait_async0();                 // this wave's buf0 writes landed
        __syncthreads();               // all waves landed; prior buf1 reads done
        async_tile_load(alds[1], blds[1],
                        ag + (it + 1) * BK, bg + (it + 1) * BK);
        compute_tile(As[0], Bs[0], acc, wm, wn, r16, hi);

        // ---- phase 1: consume buf1, DMA k+2 into buf0 under the WMMAs ----
        wait_async0();
        __syncthreads();
        if (it + 2 < NK)
            async_tile_load(alds[0], blds[0],
                            ag + (it + 2) * BK, bg + (it + 2) * BK);
        compute_tile(As[1], Bs[1], acc, wm, wn, r16, hi);
    }

    // D layout: element i of v8f is row (i + 8*hi), column r16 of the 16x16 tile
    float* obase = out + (m0 + wm + hi * 8) * OUT_DIM + n0 + wn + r16;
    #pragma unroll
    for (int mt = 0; mt < 4; ++mt) {
        #pragma unroll
        for (int nt = 0; nt < 4; ++nt) {
            float* p = obase + (size_t)(mt * 16) * OUT_DIM + nt * 16;
            #pragma unroll
            for (int i = 0; i < 8; ++i)
                p[(size_t)i * OUT_DIM] = acc[mt][nt][i];
        }
    }
}

// ---------------------------------------------------------------------------
extern "C" void kernel_launch(void* const* d_in, const int* in_sizes, int n_in,
                              void* d_out, int out_size, void* d_ws, size_t ws_size,
                              hipStream_t stream) {
    (void)in_sizes; (void)n_in; (void)out_size; (void)ws_size;
    const float* x     = (const float*)d_in[0];   // [4,2048,4096] f32
    const int*   wq    = (const int*)  d_in[1];   // [4096,4096] int32 (int8 values)
    const float* scale = (const float*)d_in[2];   // [4096*4096/64] f32
    const float* A     = (const float*)d_in[3];   // [4096,16] f32
    const float* Bl    = (const float*)d_in[4];   // [16,4096] f32
    float* out = (float*)d_out;

    _Float16* Weff = (_Float16*)d_ws;                                   // 32 MB
    _Float16* Xh   = (_Float16*)((char*)d_ws +
                       (size_t)OUT_DIM * IN_DIM * sizeof(_Float16));    // 64 MB

    lora_wconv_kernel<<<OUT_DIM, 256, 0, stream>>>(wq, scale, A, Bl, Weff);

    const int xblocks = (int)(((size_t)M_DIM * IN_DIM) / (8 * 256));    // 16384
    xconv_kernel<<<xblocks, 256, 0, stream>>>(x, Xh);

    dim3 grid(OUT_DIM / BN, M_DIM / BM);   // (16, 64)
    lora_gemm_kernel<<<grid, 256, 0, stream>>>(Xh, Weff, out);
}
